// Bottleneck_83562883711089
// MI455X (gfx1250) — compile-verified, matrix operand order checked by reading, not run on hardware
//
#include <hip/hip_runtime.h>
#include <hip/hip_bf16.h>

typedef unsigned short ushort_t;
typedef __attribute__((ext_vector_type(16))) __bf16 v16bf;
typedef __attribute__((ext_vector_type(8)))  float  v8f;
typedef __attribute__((ext_vector_type(4)))  unsigned int u32x4;

union BF16x16 { v16bf v; u32x4 q[2]; ushort_t u[16]; };

// Problem dimensions (fixed by the reference)
#define BATCH 2
#define CIN   256
#define CMID  64
#define CRED  16
#define COUT  256
#define HW    3136      // 56*56
#define HDIM  56
#define SPAN  900       // G*K*K = 4*225
#define SPANP 912       // padded to multiple of 16
#define GRP   4
#define GC    16
#define KK    225
#define NTILE 196       // 3136/16

__device__ __forceinline__ ushort_t f2bf(float f) {
    unsigned u = __float_as_uint(f);
    unsigned r = (u + 0x7FFFu + ((u >> 16) & 1u)) >> 16;   // round-to-nearest-even
    return (ushort_t)r;
}

// ---------------- prep ----------------
// x [B][256][3136] fp32 -> xbf [B][3136][256] bf16 (pixel-major, coalesced writes)
__global__ void cast_x_t_kernel(const float* __restrict__ x, ushort_t* __restrict__ xbf) {
    int o = blockIdx.x * blockDim.x + threadIdx.x;
    if (o >= BATCH * HW * CIN) return;
    const int c  = o & (CIN - 1);
    const int bp = o >> 8;               // b*HW + p
    const int b  = bp / HW;
    const int p  = bp % HW;
    xbf[o] = f2bf(x[((size_t)b * CIN + c) * HW + p]);
}

__global__ void fold_w_kernel(const float* __restrict__ W, const float* __restrict__ g,
                              ushort_t* __restrict__ Wb, int rows, int cols) {
    int i = blockIdx.x * blockDim.x + threadIdx.x;
    if (i < rows * cols) Wb[i] = f2bf(g[i / cols] * W[i]);
}

// Ws [900][16] -> Wsbf [912][16] bf16, rows 900..911 zero
__global__ void pad_ws_kernel(const float* __restrict__ Ws, ushort_t* __restrict__ Wsbf) {
    int i = blockIdx.x * blockDim.x + threadIdx.x;
    if (i >= SPANP * CRED) return;
    const int s = i >> 4;
    Wsbf[i] = (s < SPAN) ? f2bf(Ws[i]) : (ushort_t)0;
}

// ---------------- conv1: 64x256 GEMM, bf16 WMMA, bias+ReLU ----------------
// grid.x = BATCH*NTILE, block = 128 (4 waves, one 16-row m-tile each)
__global__ __launch_bounds__(128) void conv1_wmma_kernel(
    const ushort_t* __restrict__ W1bf,   // [64][256]  (g1 folded)
    const ushort_t* __restrict__ xbf,    // [B][3136][256] pixel-major
    const float*    __restrict__ b1,     // [64]
    float*          __restrict__ h1f,    // [B][64][3136]  channel-major fp32
    ushort_t*       __restrict__ h1bf)   // [B][3136][64]  pixel-major bf16
{
    const int wave = threadIdx.x >> 5;
    const int lane = threadIdx.x & 31;
    const int nt   = blockIdx.x % NTILE;
    const int b    = blockIdx.x / NTILE;
    const int n0   = nt * 16;
    const int mrow = lane & 15;
    const int kh8  = (lane >> 4) * 8;

    v8f acc = {};
    for (int k0 = 0; k0 < CIN; k0 += 32) {
        BF16x16 A, Bm;
        const ushort_t* Wp = W1bf + (size_t)(16 * wave + mrow) * CIN + k0 + kh8;
        A.q[0] = *(const u32x4*)Wp;
        A.q[1] = *(const u32x4*)(Wp + 16);
        const ushort_t* Xp = xbf + ((size_t)(b * HW + n0 + mrow)) * CIN + k0 + kh8;
        Bm.q[0] = *(const u32x4*)Xp;
        Bm.q[1] = *(const u32x4*)(Xp + 16);
        acc = __builtin_amdgcn_wmma_f32_16x16x32_bf16(false, A.v, false, Bm.v,
                                                      (short)0, acc, false, false);
    }
    const int Ncol = lane & 15;
    const int m0   = 16 * wave + kh8;       // Mbase == 8*(lane>>4) == kh8
    union { u32x4 q; ushort_t u[8]; } st;
#pragma unroll
    for (int r = 0; r < 8; ++r) {
        const int m = m0 + r;
        const float v = fmaxf(acc[r] + b1[m], 0.0f);
        h1f[((size_t)b * CMID + m) * HW + n0 + Ncol] = v;
        st.u[r] = f2bf(v);
    }
    *(u32x4*)(h1bf + ((size_t)(b * HW + n0 + Ncol)) * CMID + m0) = st.q;
}

// ---------------- kernel generation on WMMA ----------------
// reduce: rv[16xN] = relu((gr.Wr)[16x64] @ h1[64xN] + br)   (2 WMMAs)
// span:   w[912xN] = Ws[912x16] @ rv[16xN] + bs             (57 WMMAs, K zero-padded to 32)
// C/D layout of reduce maps in-lane onto B layout of span -> no shuffles.
// grid.x = BATCH*NTILE/4 = 98, block = 128 (one 16-pixel n-tile per wave)
__global__ __launch_bounds__(128) void kgen_wmma_kernel(
    const ushort_t* __restrict__ Wrbf,   // [16][64] (gr folded)
    const ushort_t* __restrict__ h1bf,   // [B][3136][64] pixel-major
    const float*    __restrict__ br,     // [16]
    const ushort_t* __restrict__ Wsbf,   // [912][16] zero-padded
    const float*    __restrict__ bs,     // [900]
    float*          __restrict__ wbuf)   // [B][912][3136]
{
    const int wave = threadIdx.x >> 5;
    const int lane = threadIdx.x & 31;
    const int idx  = blockIdx.x * 4 + wave;
    const int b    = idx / NTILE;
    const int nt   = idx % NTILE;
    const int n0   = nt * 16;
    const int col  = lane & 15;
    const int kh8  = (lane >> 4) * 8;

    v8f racc = {};
    for (int k0 = 0; k0 < CMID; k0 += 32) {
        BF16x16 A, Bm;
        const ushort_t* Wp = Wrbf + (size_t)col * CMID + k0 + kh8;
        A.q[0] = *(const u32x4*)Wp;
        A.q[1] = *(const u32x4*)(Wp + 16);
        const ushort_t* Hp = h1bf + ((size_t)(b * HW + n0 + col)) * CMID + k0 + kh8;
        Bm.q[0] = *(const u32x4*)Hp;
        Bm.q[1] = *(const u32x4*)(Hp + 16);
        racc = __builtin_amdgcn_wmma_f32_16x16x32_bf16(false, A.v, false, Bm.v,
                                                       (short)0, racc, false, false);
    }
    // rv (bias + ReLU) directly into span's B-operand layout, upper K half = 0
    BF16x16 R;
#pragma unroll
    for (int e = 0; e < 8; ++e)
        R.u[e] = f2bf(fmaxf(racc[e] + br[e + kh8], 0.0f));
    {
        u32x4 z = {};
        R.q[1] = z;
    }

    for (int mt = 0; mt < SPANP / 16; ++mt) {
        const int m0 = mt * 16;
        BF16x16 A;
        A.q[0] = *(const u32x4*)(Wsbf + (size_t)(m0 + col) * CRED + kh8);
        {
            u32x4 z = {};
            A.q[1] = z;
        }
        v8f acc = {};
        acc = __builtin_amdgcn_wmma_f32_16x16x32_bf16(false, A.v, false, R.v,
                                                      (short)0, acc, false, false);
#pragma unroll
        for (int r = 0; r < 8; ++r) {
            const int s = m0 + kh8 + r;
            if (s < SPAN)
                wbuf[((size_t)b * SPANP + s) * HW + n0 + col] = acc[r] + bs[s];
        }
    }
}

// ---------------- involution + BN + ReLU -> bf16 (pixel-major) ----------------
// grid = (49 tiles, 4 groups, 2 batch), block = 64 threads (one 8x8 pixel tile)
__global__ __launch_bounds__(64) void invol_kernel(
    const float* __restrict__ h1f,   // [B][64][3136]
    const float* __restrict__ wbuf,  // [B][912][3136], s = g*225+k
    const float* __restrict__ g2, const float* __restrict__ b2,
    ushort_t*    __restrict__ h2bf)  // [B][3136][64] pixel-major
{
    __shared__ float sh[GC * 484];   // 16 channels x 22x22 halo region = 31 KB

    const int tile = blockIdx.x;
    const int g    = blockIdx.y;
    const int b    = blockIdx.z;
    const int y0 = (tile / 7) * 8;
    const int x0 = (tile % 7) * 8;

    const float* hg = h1f + ((size_t)b * CMID + g * GC) * HW;
    for (int i = threadIdx.x; i < GC * 484; i += 64) {
        const int c  = i / 484;
        const int rr = i % 484;
        const int gy = y0 - 7 + rr / 22;
        const int gx = x0 - 7 + rr % 22;
        float v = 0.0f;
        if (gy >= 0 && gy < HDIM && gx >= 0 && gx < HDIM)
            v = hg[(size_t)c * HW + gy * HDIM + gx];
        sh[i] = v;
    }
    __syncthreads();

    const int py = threadIdx.x / 8;
    const int px = threadIdx.x % 8;
    float acc[GC];
#pragma unroll
    for (int c = 0; c < GC; ++c) acc[c] = 0.0f;

    const float* wp = wbuf + ((size_t)b * SPANP + g * KK) * HW + (y0 + py) * HDIM + (x0 + px);
    for (int k = 0; k < KK; ++k) {
        const float wk = wp[(size_t)k * HW];
        const int idx2 = (py + k / 15) * 22 + (px + k % 15);
#pragma unroll
        for (int c = 0; c < GC; ++c) acc[c] += wk * sh[c * 484 + idx2];
    }

    const int p = (y0 + py) * HDIM + (x0 + px);
    union { u32x4 q[2]; ushort_t u[16]; } st;
#pragma unroll
    for (int c = 0; c < GC; ++c) {
        const int cg = g * GC + c;
        st.u[c] = f2bf(fmaxf(g2[cg] * acc[c] + b2[cg], 0.0f));
    }
    ushort_t* hp = h2bf + ((size_t)(b * HW + p)) * CMID + g * GC;
    *(u32x4*)(hp)     = st.q[0];
    *(u32x4*)(hp + 8) = st.q[1];
}

// ---------------- conv3: 256x64 GEMM, bf16 WMMA, bias + residual + ReLU ----------------
// grid = (BATCH*NTILE, 4), block = 128 (4 waves -> 64 output rows per block)
__global__ __launch_bounds__(128) void conv3_wmma_kernel(
    const ushort_t* __restrict__ W3bf,  // [256][64] (g3 folded)
    const ushort_t* __restrict__ h2bf,  // [B][3136][64] pixel-major
    const float*    __restrict__ b3,    // [256]
    const float*    __restrict__ xres,  // [B][256][3136]
    float*          __restrict__ out)   // [B][256][3136]
{
    const int wave = threadIdx.x >> 5;
    const int lane = threadIdx.x & 31;
    const int nt   = blockIdx.x % NTILE;
    const int b    = blockIdx.x / NTILE;
    const int n0   = nt * 16;
    const int m0   = 64 * blockIdx.y + 16 * wave;
    const int mrow = lane & 15;
    const int kh8  = (lane >> 4) * 8;

    v8f acc = {};
    for (int k0 = 0; k0 < CMID; k0 += 32) {
        BF16x16 A, Bm;
        const ushort_t* Wp = W3bf + (size_t)(m0 + mrow) * CMID + k0 + kh8;
        A.q[0] = *(const u32x4*)Wp;
        A.q[1] = *(const u32x4*)(Wp + 16);
        const ushort_t* Hp = h2bf + ((size_t)(b * HW + n0 + mrow)) * CMID + k0 + kh8;
        Bm.q[0] = *(const u32x4*)Hp;
        Bm.q[1] = *(const u32x4*)(Hp + 16);
        acc = __builtin_amdgcn_wmma_f32_16x16x32_bf16(false, A.v, false, Bm.v,
                                                      (short)0, acc, false, false);
    }
    const int Ncol = lane & 15;
#pragma unroll
    for (int r = 0; r < 8; ++r) {
        const int o = m0 + kh8 + r;
        const size_t off = ((size_t)b * COUT + o) * HW + n0 + Ncol;
        out[off] = fmaxf(acc[r] + b3[o] + xres[off], 0.0f);
    }
}

extern "C" void kernel_launch(void* const* d_in, const int* in_sizes, int n_in,
                              void* d_out, int out_size, void* d_ws, size_t ws_size,
                              hipStream_t stream) {
    const float* x  = (const float*)d_in[0];
    const float* W1 = (const float*)d_in[1];
    const float* g1 = (const float*)d_in[2];
    const float* b1 = (const float*)d_in[3];
    const float* Wr = (const float*)d_in[4];
    const float* gr = (const float*)d_in[5];
    const float* br = (const float*)d_in[6];
    const float* Ws = (const float*)d_in[7];
    const float* bs = (const float*)d_in[8];
    const float* g2 = (const float*)d_in[9];
    const float* b2 = (const float*)d_in[10];
    const float* W3 = (const float*)d_in[11];
    const float* g3 = (const float*)d_in[12];
    const float* b3 = (const float*)d_in[13];
    float* out = (float*)d_out;

    char* ws = (char*)d_ws;
    size_t off = 0;
    auto carve = [&](size_t bytes) -> void* {
        void* p = ws + off;
        off = (off + bytes + 255) & ~(size_t)255;
        return p;
    };
    ushort_t* xbf  = (ushort_t*)carve((size_t)BATCH * HW * CIN * sizeof(ushort_t));
    ushort_t* W1bf = (ushort_t*)carve((size_t)CMID * CIN * sizeof(ushort_t));
    ushort_t* W3bf = (ushort_t*)carve((size_t)COUT * CMID * sizeof(ushort_t));
    ushort_t* Wrbf = (ushort_t*)carve((size_t)CRED * CMID * sizeof(ushort_t));
    ushort_t* Wsbf = (ushort_t*)carve((size_t)SPANP * CRED * sizeof(ushort_t));
    float*    h1f  = (float*)   carve((size_t)BATCH * CMID * HW * sizeof(float));
    ushort_t* h1bf = (ushort_t*)carve((size_t)BATCH * HW * CMID * sizeof(ushort_t));
    float*    wbuf = (float*)   carve((size_t)BATCH * SPANP * HW * sizeof(float));
    ushort_t* h2bf = (ushort_t*)carve((size_t)BATCH * HW * CMID * sizeof(ushort_t));
    (void)ws_size; (void)in_sizes; (void)n_in; (void)out_size;

    // prep: transpose/cast x, fold BN scales into bf16 weights, pad span weights
    {
        const int n = BATCH * HW * CIN;
        cast_x_t_kernel<<<(n + 255) / 256, 256, 0, stream>>>(x, xbf);
        fold_w_kernel<<<(CMID * CIN + 255) / 256, 256, 0, stream>>>(W1, g1, W1bf, CMID, CIN);
        fold_w_kernel<<<(COUT * CMID + 255) / 256, 256, 0, stream>>>(W3, g3, W3bf, COUT, CMID);
        fold_w_kernel<<<(CRED * CMID + 255) / 256, 256, 0, stream>>>(Wr, gr, Wrbf, CRED, CMID);
        pad_ws_kernel<<<(SPANP * CRED + 255) / 256, 256, 0, stream>>>(Ws, Wsbf);
    }
    // conv1 (WMMA) + BN + ReLU -> h1f (fp32 cmaj) + h1bf (bf16 pmaj)
    conv1_wmma_kernel<<<dim3(BATCH * NTILE), 128, 0, stream>>>(W1bf, xbf, b1, h1f, h1bf);
    // involution kernel generation (WMMA reduce + WMMA span)
    kgen_wmma_kernel<<<dim3(BATCH * NTILE / 4), 128, 0, stream>>>(Wrbf, h1bf, br, Wsbf, bs, wbuf);
    // involution gather + BN + ReLU -> bf16 pixel-major
    invol_kernel<<<dim3(49, GRP, BATCH), 64, 0, stream>>>(h1f, wbuf, g2, b2, h2bf);
    // conv3 (WMMA) + BN + residual + ReLU
    conv3_wmma_kernel<<<dim3(BATCH * NTILE, 4), 128, 0, stream>>>(W3bf, h2bf, b3, x, out);
}
